// ReprogrammingLayer_76587856822991
// MI455X (gfx1250) — matrix operand order repaired and checked by reading, not compiled, not run on hardware
//
#include <hip/hip_runtime.h>
#include <hip/hip_bf16.h>
#include <math.h>

// ---------------- CDNA5 WMMA types & helpers ----------------
typedef __attribute__((ext_vector_type(16))) __bf16 v16bf;
typedef __attribute__((ext_vector_type(8)))  float  v8f;

#define NH     16
#define DK     64
#define HD     (NH * DK)      // 1024
#define LSEQ   2048
#define BATCH  8
#define SVALID 1000
#define SPAD   1024

static __device__ inline v8f wmma_bf16(v16bf a, v16bf b, v8f c) {
  return __builtin_amdgcn_wmma_f32_16x16x32_bf16(false, a, false, b, (short)0, c,
                                                 false, false);
}

// Async global->LDS 16-byte copy (CDNA5 GLOBAL_LOAD_ASYNC_TO_LDS_B128, ASYNCcnt).
// ldsOff = LDS byte address (low 32 bits of the generic shared pointer).
static __device__ inline void async_g2l_b128(uint32_t ldsOff, const __bf16* g) {
  asm volatile("global_load_async_to_lds_b128 %0, %1, off"
               :: "v"(ldsOff), "v"((unsigned long long)(uintptr_t)g)
               : "memory");
}

// A fragment: 16x32 bf16 from row-major source (pre-offset), stride lda (multiple of 8).
// ISA layout: lane<16 -> M=lane, elements 0..7 <-> k=half*8..+7, 8..15 <-> k=16+half*8..+7.
static __device__ inline v16bf load_a_frag(const __bf16* A, int lda, int lane) {
  const int half = lane >> 4;
  const int m    = lane & 15;
  const __bf16* p = A + (size_t)m * lda + half * 8;
  union { v16bf v; uint4 q[2]; } f;
  f.q[0] = *(const uint4*)(p);
  f.q[1] = *(const uint4*)(p + 16);
  return f.v;
}

// B fragment from TRANSPOSED storage Bt[n][k] (logical B[k][n]), stride ldt (mult of 8).
// Elements 0..7 <-> k = half*8..+7, 8..15 <-> k = 16+half*8..+7 : two b128 loads.
static __device__ inline v16bf load_bt_frag(const __bf16* Bt, int ldt, int lane) {
  const int half = lane >> 4;
  const int n    = lane & 15;
  const __bf16* p = Bt + (size_t)n * ldt + half * 8;
  union { v16bf v; uint4 q[2]; } f;
  f.q[0] = *(const uint4*)(p);
  f.q[1] = *(const uint4*)(p + 16);
  return f.v;
}

// ---------------- fp32 -> bf16 converts ----------------
__global__ __launch_bounds__(256) void cvt_f32_bf16(const float* __restrict__ in,
                                                    __bf16* __restrict__ out, int n) {
  int i = blockIdx.x * blockDim.x + threadIdx.x;
  int stride = gridDim.x * blockDim.x;
  for (; i < n; i += stride) out[i] = (__bf16)in[i];
}

__global__ __launch_bounds__(256) void cvt_f32_bf16_pad(const float* __restrict__ in,
                                                        __bf16* __restrict__ out,
                                                        int nvalid, int ntotal) {
  int i = blockIdx.x * blockDim.x + threadIdx.x;
  int stride = gridDim.x * blockDim.x;
  for (; i < ntotal; i += stride) out[i] = (i < nvalid) ? (__bf16)in[i] : (__bf16)0.0f;
}

// LDS-tiled transpose + convert: in f32 [K][N] -> out bf16 [N][K]. Block (32,8).
__global__ __launch_bounds__(256) void cvt_transpose(const float* __restrict__ in,
                                                     __bf16* __restrict__ out,
                                                     int K, int N) {
  __shared__ float tile[32][33];
  const int k0 = blockIdx.y * 32, n0 = blockIdx.x * 32;
  const int tx = threadIdx.x, ty = threadIdx.y;
#pragma unroll
  for (int j = 0; j < 32; j += 8)
    tile[ty + j][tx] = in[(size_t)(k0 + ty + j) * N + n0 + tx];
  __syncthreads();
#pragma unroll
  for (int j = 0; j < 32; j += 8)
    out[(size_t)(n0 + ty + j) * K + k0 + tx] = (__bf16)tile[tx][ty + j];
}

// ---------------- bf16 WMMA GEMM: out = A[M,K] @ Bt[N,K]^T + bias ----------------
// Block: 256 thr (8 waves) -> 128x64 tile; wave (mw,nw) computes 32x32 (4 WMMA/k-step).
// Double-buffered LDS fed by async global->LDS B128 copies (ASYNCcnt pipelining):
// issue next tile into buf^1, s_wait_asynccnt 3 guarantees current tile resident
// (ASYNCcnt completes in issue order), compute, barrier, swap.
__global__ __launch_bounds__(256)
void gemm_bf16(const __bf16* __restrict__ A, const __bf16* __restrict__ Bt,
               const float* __restrict__ bias, __bf16* __restrict__ outB,
               float* __restrict__ outF, int N, int K, int Mvalid,
               int ldOut, int transOut) {
  __shared__ __align__(16) __bf16 As[2][128][40];
  __shared__ __align__(16) __bf16 Bs[2][64][40];
  const int tid  = threadIdx.x;
  const int lane = tid & 31;
  const int wave = tid >> 5;
  const int mw   = wave & 3;   // 4 M-subtiles of 32
  const int nw   = wave >> 2;  // 2 N-subtiles of 32
  const int m0   = blockIdx.y * 128;
  const int n0   = blockIdx.x * 64;

  // per-thread staging coordinates (2 A-chunks + 1 B-chunk of 16B each)
  const int ra0 = tid >> 2,          ka0 = (tid & 3) * 8;
  const int ra1 = (tid + 256) >> 2,  ka1 = ka0;             // (tid+256)&3 == tid&3
  const int rb  = tid >> 2,          kb  = ka0;

  auto issue = [&](int buf, int k0) {
    async_g2l_b128((uint32_t)(uintptr_t)&As[buf][ra0][ka0],
                   &A[(size_t)(m0 + ra0) * K + k0 + ka0]);
    async_g2l_b128((uint32_t)(uintptr_t)&As[buf][ra1][ka1],
                   &A[(size_t)(m0 + ra1) * K + k0 + ka1]);
    async_g2l_b128((uint32_t)(uintptr_t)&Bs[buf][rb][kb],
                   &Bt[(size_t)(n0 + rb) * K + k0 + kb]);
  };

  v8f c00 = {}, c01 = {}, c10 = {}, c11 = {};

  issue(0, 0);
  int buf = 0;
  for (int k0 = 0; k0 < K; k0 += 32) {
    if (k0 + 32 < K) {
      issue(buf ^ 1, k0 + 32);
      asm volatile("s_wait_asynccnt 0x3" ::: "memory");  // current tile resident
    } else {
      asm volatile("s_wait_asynccnt 0x0" ::: "memory");
    }
    __syncthreads();
    v16bf a0 = load_a_frag(&As[buf][mw * 32][0], 40, lane);
    v16bf a1 = load_a_frag(&As[buf][mw * 32 + 16][0], 40, lane);
    v16bf b0 = load_bt_frag(&Bs[buf][nw * 32][0], 40, lane);
    v16bf b1 = load_bt_frag(&Bs[buf][nw * 32 + 16][0], 40, lane);
    c00 = wmma_bf16(a0, b0, c00);
    c01 = wmma_bf16(a0, b1, c01);
    c10 = wmma_bf16(a1, b0, c10);
    c11 = wmma_bf16(a1, b1, c11);
    __syncthreads();  // all waves done reading buf before it is overwritten
    buf ^= 1;
  }

  const int half = lane >> 4;
  const int n    = lane & 15;
#pragma unroll
  for (int mi = 0; mi < 2; ++mi) {
    v8f cx = mi ? c10 : c00;
    v8f cy = mi ? c11 : c01;
#pragma unroll
    for (int i = 0; i < 8; ++i) {
      int   row  = m0 + mw * 32 + mi * 16 + i + half * 8;
      int   col0 = n0 + nw * 32 + n;
      int   col1 = col0 + 16;
      float v0   = (row < Mvalid) ? (cx[i] + bias[col0]) : 0.0f;
      float v1   = (row < Mvalid) ? (cy[i] + bias[col1]) : 0.0f;
      size_t i0 = transOut ? ((size_t)col0 * ldOut + row) : ((size_t)row * ldOut + col0);
      size_t i1 = transOut ? ((size_t)col1 * ldOut + row) : ((size_t)row * ldOut + col1);
      if (outF) { outF[i0] = v0;          outF[i1] = v1; }
      else      { outB[i0] = (__bf16)v0;  outB[i1] = (__bf16)v1; }
    }
  }
}

// ---------------- fused flash cross-attention ----------------
// One wave owns 16 query rows of one (b,h); streams S in steps of 32, online softmax.
// K is row-major [SPAD][HD] (natural Bt for Q@K^T); V is transposed [HD][SPAD].
__global__ __launch_bounds__(256)
void flash_attn(const __bf16* __restrict__ Q, const __bf16* __restrict__ Kr,
                const __bf16* __restrict__ Vt, __bf16* __restrict__ O) {
  __shared__ __align__(16) __bf16 Ps[8][16 * 40];
  const int lane = threadIdx.x & 31;
  const int wiv  = threadIdx.x >> 5;
  const int gw   = blockIdx.x * 8 + wiv;
  const int nLt  = LSEQ / 16;  // 128
  const int lt   = gw % nLt;
  int       t2   = gw / nLt;
  const int h    = t2 % NH;
  const int b    = t2 / NH;

  const __bf16* Qp = Q + ((size_t)(b * LSEQ + lt * 16)) * HD + h * DK;
  v16bf aq0 = load_a_frag(Qp, HD, lane);       // dk 0..31
  v16bf aq1 = load_a_frag(Qp + 32, HD, lane);  // dk 32..63

  float mrun[8], lrun[8];
  v8f   acc[4] = {};
#pragma unroll
  for (int i = 0; i < 8; ++i) { mrun[i] = -1e30f; lrun[i] = 0.0f; }

  const int half = lane >> 4;
  const int n    = lane & 15;
  __bf16*   myP  = &Ps[wiv][0];
  const float scale = 0.125f;  // 1/sqrt(64)

  for (int s0 = 0; s0 < SPAD; s0 += 32) {
    // scores: two 16x16 tiles (cols s0..s0+31); dk = 64 -> 2 WMMA each
    v8f c0 = {}, c1 = {};
    const __bf16* kt0 = Kr + (size_t)s0 * HD + h * DK;
    c0 = wmma_bf16(aq0, load_bt_frag(kt0, HD, lane), c0);
    c0 = wmma_bf16(aq1, load_bt_frag(kt0 + 32, HD, lane), c0);
    const __bf16* kt1 = kt0 + (size_t)16 * HD;
    c1 = wmma_bf16(aq0, load_bt_frag(kt1, HD, lane), c1);
    c1 = wmma_bf16(aq1, load_bt_frag(kt1 + 32, HD, lane), c1);

    const bool ok0 = (s0 + n) < SVALID;
    const bool ok1 = (s0 + 16 + n) < SVALID;

    float p0[8], p1[8];
#pragma unroll
    for (int i = 0; i < 8; ++i) {
      float s0v = ok0 ? c0[i] * scale : -1e30f;
      float s1v = ok1 ? c1[i] * scale : -1e30f;
      float rm  = fmaxf(s0v, s1v);
#pragma unroll
      for (int off = 1; off < 16; off <<= 1) rm = fmaxf(rm, __shfl_xor(rm, off, 16));
      float mnew = fmaxf(mrun[i], rm);
      float fac  = __expf(mrun[i] - mnew);
      p0[i]      = __expf(s0v - mnew);
      p1[i]      = __expf(s1v - mnew);
      float rs   = p0[i] + p1[i];
#pragma unroll
      for (int off = 1; off < 16; off <<= 1) rs += __shfl_xor(rs, off, 16);
      lrun[i] = lrun[i] * fac + rs;
      mrun[i] = mnew;
#pragma unroll
      for (int t = 0; t < 4; ++t) acc[t][i] *= fac;
    }
    // restage P (C layout -> A layout) via per-wave LDS; same-wave DS ops are in-order
#pragma unroll
    for (int i = 0; i < 8; ++i) {
      int m                = i + half * 8;
      myP[m * 40 + n]      = (__bf16)p0[i];
      myP[m * 40 + 16 + n] = (__bf16)p1[i];
    }
    v16bf ap = load_a_frag(myP, 40, lane);
#pragma unroll
    for (int t = 0; t < 4; ++t) {
      const __bf16* vt = Vt + (size_t)(h * DK + t * 16) * SPAD + s0;
      acc[t] = wmma_bf16(ap, load_bt_frag(vt, SPAD, lane), acc[t]);
    }
  }

  __bf16* Op = O + ((size_t)(b * LSEQ + lt * 16)) * HD + h * DK;
#pragma unroll
  for (int t = 0; t < 4; ++t) {
#pragma unroll
    for (int i = 0; i < 8; ++i) {
      int m                           = i + half * 8;
      Op[(size_t)m * HD + t * 16 + n] = (__bf16)(acc[t][i] / lrun[i]);
    }
  }
}

// ---------------- host orchestration ----------------
extern "C" void kernel_launch(void* const* d_in, const int* in_sizes, int n_in,
                              void* d_out, int out_size, void* d_ws, size_t ws_size,
                              hipStream_t stream) {
  (void)in_sizes; (void)n_in; (void)out_size; (void)ws_size;
  const float* X  = (const float*)d_in[0];   // [8,2048,1024]
  const float* SK = (const float*)d_in[1];   // [1000,4096]
  const float* SV = (const float*)d_in[2];   // [1000,4096]
  const float* Wq = (const float*)d_in[3];   // [1024,1024]
  const float* bq = (const float*)d_in[4];
  const float* Wk = (const float*)d_in[5];   // [4096,1024]
  const float* bk = (const float*)d_in[6];
  const float* Wv = (const float*)d_in[7];   // [4096,1024]
  const float* bv = (const float*)d_in[8];
  const float* Wo = (const float*)d_in[9];   // [1024,4096]
  const float* bo = (const float*)d_in[10];
  float*       out = (float*)d_out;          // [16384,4096] fp32

  char*  ws  = (char*)d_ws;
  size_t off = 0;
  auto take  = [&](size_t elems) {
    __bf16* p = (__bf16*)(ws + off);
    off += ((elems * sizeof(__bf16)) + 255) & ~(size_t)255;
    return p;
  };
  const size_t nX = (size_t)BATCH * LSEQ * HD;  // 16,777,216
  __bf16* Xb   = take(nX);                       // A for Q proj (reused as Ob)
  __bf16* WqT  = take((size_t)HD * HD);          // Wq^T  [1024][1024]
  __bf16* WkT  = take((size_t)HD * 4096);        // Wk^T  [1024][4096]
  __bf16* WvT  = take((size_t)HD * 4096);        // Wv^T  [1024][4096]
  __bf16* WoT  = take((size_t)4096 * HD);        // Wo^T  [4096][1024]
  __bf16* SKb  = take((size_t)SPAD * 4096);      // zero-padded to 1024 rows
  __bf16* SVb  = take((size_t)SPAD * 4096);
  __bf16* Qb   = take(nX);
  __bf16* Kbp  = take((size_t)SPAD * HD);        // K rows  [s][hd]
  __bf16* Vtp  = take((size_t)HD * SPAD);        // V^T     [hd][s]
  __bf16* Ob   = Xb;  // X dead after Q projection (stream-ordered reuse)

  // converts
  cvt_f32_bf16<<<2048, 256, 0, stream>>>(X, Xb, (int)nX);
  cvt_f32_bf16_pad<<<2048, 256, 0, stream>>>(SK, SKb, SVALID * 4096, SPAD * 4096);
  cvt_f32_bf16_pad<<<2048, 256, 0, stream>>>(SV, SVb, SVALID * 4096, SPAD * 4096);
  // weight transpose-converts: in [K][N] -> out [N][K]
  cvt_transpose<<<dim3(HD / 32, HD / 32), dim3(32, 8), 0, stream>>>(Wq, WqT, HD, HD);
  cvt_transpose<<<dim3(HD / 32, 4096 / 32), dim3(32, 8), 0, stream>>>(Wk, WkT, 4096, HD);
  cvt_transpose<<<dim3(HD / 32, 4096 / 32), dim3(32, 8), 0, stream>>>(Wv, WvT, 4096, HD);
  cvt_transpose<<<dim3(4096 / 32, HD / 32), dim3(32, 8), 0, stream>>>(Wo, WoT, HD, 4096);

  // K projection: [1024,4096] @ WkT -> Kbp row-major [s][hd], padded rows zeroed
  gemm_bf16<<<dim3(HD / 64, SPAD / 128), 256, 0, stream>>>(
      SKb, WkT, bk, Kbp, nullptr, HD, 4096, SVALID, HD, 0);
  // V projection: -> Vtp transposed [hd][s]
  gemm_bf16<<<dim3(HD / 64, SPAD / 128), 256, 0, stream>>>(
      SVb, WvT, bv, Vtp, nullptr, HD, 4096, SVALID, SPAD, 1);
  // Q projection: [16384,1024] @ WqT -> Qb
  gemm_bf16<<<dim3(HD / 64, (BATCH * LSEQ) / 128), 256, 0, stream>>>(
      Xb, WqT, bq, Qb, nullptr, HD, HD, BATCH * LSEQ, HD, 0);
  // fused attention: B*H*(L/16) = 16384 waves, 8 waves/block
  flash_attn<<<(BATCH * NH * (LSEQ / 16)) / 8, 256, 0, stream>>>(Qb, Kbp, Vtp, Ob);
  // output projection: [16384,1024] @ WoT + bo -> fp32 d_out
  gemm_bf16<<<dim3(4096 / 64, (BATCH * LSEQ) / 128), 256, 0, stream>>>(
      Ob, WoT, bo, nullptr, out, 4096, HD, BATCH * LSEQ, 4096, 0);
}
